// VelocityCorrector_30872224924183
// MI455X (gfx1250) — matrix operand order repaired and checked by reading, not compiled, not run on hardware
//
#include <hip/hip_runtime.h>

// ---------------- problem dims ----------------
#define BDIM 4096
#define CDIM 128
#define VDIM 64
#define HDIM 3
#define K2C  256       // 2*C
#define NB   4         // batches per workgroup
#define NCOL (NB*VDIM) // 256 columns in big GEMM

// ---------------- ws (scratch) layout, bf16 elements ----------------
#define WSTK_OFF  0
#define WSTK_ELTS (8*8*16*256)              // 262144: [ctile][block][16][256]
#define WC1_OFF   (WSTK_OFF + WSTK_ELTS)    // 262144
#define WC2_OFF   (WC1_OFF + CDIM*CDIM)     // +16384
#define ATB_OFF   (WC2_OFF + CDIM*CDIM)     // +16384
#define WS_ELTS   (ATB_OFF + HDIM*VDIM*VDIM)// +12288 = 307200

// ---------------- LDS layout (bytes), strides in bf16 elements ----------------
#define XP 264  // x buffer c-stride   (row = 528 B, 16B aligned, bank-spread)
#define AP 72   // A^T v-stride        (row = 144 B)
#define PP 72   // p/q v-stride        (row = 144 B)
#define FP 136  // fused/h c-stride    (row = 272 B)
#define XS_OFF   0
#define XS_BYTES (NB*VDIM*XP*2)             // 135168
#define AS_OFF   (XS_OFF + XS_BYTES)
#define AS_BYTES (HDIM*64*AP*2)             // 27648
#define PQ_OFF   (AS_OFF + AS_BYTES)
#define PQ_BYTES (6*16*NB*PP*2)             // 55296
#define GR_OFF   (PQ_OFF + PQ_BYTES)
#define GR_BYTES (2*16*NCOL*2)              // 16384
#define FS_OFF   (GR_OFF + GR_BYTES)
#define FS_BYTES (NB*VDIM*FP*2)             // 69632
#define SMEM_BYTES (FS_OFF + FS_BYTES)      // 304128 (< 320 KB)
#define HS_OFF   XS_OFF                     // stage-3 buffer overlays dead X

typedef __attribute__((ext_vector_type(16))) __bf16        bf16x16;
typedef __attribute__((ext_vector_type(8)))  float         floatx8;
typedef __attribute__((ext_vector_type(4)))  float         floatx4;
typedef __attribute__((ext_vector_type(4)))  unsigned int  uintx4;

union Frag { bf16x16 v; uintx4 q[2]; };

// Native hardware bf16 converts (RNE) instead of integer emulation.
__device__ __forceinline__ unsigned short f2bf(float f) {
  union { __bf16 h; unsigned short s; } u;
  u.h = (__bf16)f;
  return u.s;
}
__device__ __forceinline__ float bf2f(unsigned short s) {
  union { unsigned short s; __bf16 h; } u;
  u.s = s;
  return (float)u.h;
}
__device__ __forceinline__ floatx8 wmma_bf16(const Frag& a, const Frag& b, floatx8 c) {
  return __builtin_amdgcn_wmma_f32_16x16x32_bf16(false, a.v, false, b.v, (short)0, c,
                                                 false, false);
}

// ---------------- prep: f32 weights -> bf16 in ws, A transposed ----------------
__global__ void vc_prep(const float* __restrict__ W1, const float* __restrict__ W2,
                        const float* __restrict__ Wg, const float* __restrict__ Wr,
                        const float* __restrict__ Wc1, const float* __restrict__ Wc2,
                        const float* __restrict__ A,  unsigned short* __restrict__ ws)
{
  const int idx = blockIdx.x * 256 + threadIdx.x;
  if (idx < WSTK_ELTS) {
    const int k = idx & 255;
    const int r = (idx >> 8) & 15;
    const int j = (idx >> 12) & 7;   // 0-2: W1 heads, 3-5: W2 heads, 6: Wg, 7: Wres
    const int t = idx >> 15;         // c-tile
    const int o = t*16 + r;
    float v;
    if (j < 3)       v = W1[((size_t)j*CDIM + o)*K2C + k];
    else if (j < 6)  v = W2[((size_t)(j-3)*CDIM + o)*K2C + k];
    else if (j == 6) v = Wg[(size_t)o*K2C + k];
    else             v = Wr[(size_t)o*K2C + k];
    ws[idx] = f2bf(v);
  } else if (idx < WC2_OFF) {
    ws[idx] = f2bf(Wc1[idx - WC1_OFF]);
  } else if (idx < ATB_OFF) {
    ws[idx] = f2bf(Wc2[idx - WC2_OFF]);
  } else if (idx < WS_ELTS) {
    const int i = idx - ATB_OFF;         // dst: [h][w][v]
    const int v = i & 63;
    const int w = (i >> 6) & 63;
    const int h = i >> 12;
    ws[idx] = f2bf(A[((size_t)h*VDIM + v)*VDIM + w]);  // src A[h][v][w]
  }
}

// ---------------- main fused kernel ----------------
__global__ __launch_bounds__(256)
void vc_main(const float* __restrict__ z, const float* __restrict__ ve,
             const unsigned short* __restrict__ ws, const float* __restrict__ scale,
             float* __restrict__ out)
{
  extern __shared__ char smem[];
  unsigned short* XS = (unsigned short*)(smem + XS_OFF); // [nb*64+v][c(XP)]  bf16 x^T
  unsigned short* AS = (unsigned short*)(smem + AS_OFF); // [h][w][v(AP)]     bf16 A^T
  unsigned short* PQ = (unsigned short*)(smem + PQ_OFF); // [blk][r][nb][v(PP)]
  unsigned short* GR = (unsigned short*)(smem + GR_OFF); // [g|r][r][col]
  unsigned short* FS = (unsigned short*)(smem + FS_OFF); // [nb*64+v][c(FP)]  fused^T
  unsigned short* HS = (unsigned short*)(smem + HS_OFF); // [col][o(FP)]      h^T

  const unsigned short* wstk = ws + WSTK_OFF;
  const unsigned short* wc1b = ws + WC1_OFF;
  const unsigned short* wc2b = ws + WC2_OFF;
  const unsigned short* atb  = ws + ATB_OFF;

  const int tid  = threadIdx.x;
  const int wv   = tid >> 5;
  const int lane = tid & 31;
  const int lm   = lane & 15;
  const int hi   = lane >> 4;
  const int b0   = blockIdx.x * NB;

  // ---- Phase A: stage x (f32 -> bf16, transposed) via b128 loads ----
  {
    const int l16 = tid & 15;        // 16 lanes cover the 64-float v row as float4
    const int rg  = tid >> 4;        // 16 (nb,c) rows in flight
    for (int row = rg; row < NB*K2C; row += 16) {
      const int nb = row >> 8;
      const int c  = row & 255;
      const float* srcrow = (c < CDIM)
        ? (z  + ((size_t)(b0+nb)*CDIM + c)*VDIM)
        : (ve + ((size_t)(b0+nb)*CDIM + (c-CDIM))*VDIM);
      const floatx4 xv = *(const floatx4*)(srcrow + l16*4);
      unsigned short* dst = XS + (nb*VDIM + l16*4)*XP + c;
      dst[0*XP] = f2bf(xv.x);
      dst[1*XP] = f2bf(xv.y);
      dst[2*XP] = f2bf(xv.z);
      dst[3*XP] = f2bf(xv.w);
    }
    // A^T: vectorized b128 copy global->LDS (rows of 64 bf16 = 8 uint4 chunks)
    for (int ch = tid; ch < HDIM*64*8; ch += 256) {
      const int rowi = ch >> 3;      // [h*64+w]
      const int part = ch & 7;
      *(uintx4*)(AS + rowi*AP + part*8) = *(const uintx4*)(atb + rowi*64 + part*8);
    }
  }
  __syncthreads();

  const floatx8 vzero = {0.f,0.f,0.f,0.f,0.f,0.f,0.f,0.f};

  // ---- Phase B: loop over 8 c-tile groups ----
  for (int t = 0; t < 8; ++t) {
    // GEMM1: wave wv computes its block (p/q/g/r) rows, K=256, N=256
    floatx8 acc[16];
    #pragma unroll
    for (int n = 0; n < 16; ++n) acc[n] = vzero;
    for (int kt = 0; kt < 8; ++kt) {
      Frag a;
      const unsigned short* wrow = wstk + (((t*8 + wv)*16 + lm)*K2C) + kt*32 + hi*8;
      a.q[0] = *(const uintx4*)(wrow);
      a.q[1] = *(const uintx4*)(wrow + 16);
      #pragma unroll
      for (int n = 0; n < 16; ++n) {
        Frag b;
        const unsigned short* xrow = XS + (n*16 + lm)*XP + kt*32 + hi*8;
        b.q[0] = *(const uintx4*)(xrow);
        b.q[1] = *(const uintx4*)(xrow + 16);
        acc[n] = wmma_bf16(a, b, acc[n]);
      }
    }
    if (wv < 6) {               // p_h / q_h -> PQ staging (A-operand layout)
      #pragma unroll
      for (int n = 0; n < 16; ++n) {
        const int col = n*16 + lm, nb = col >> 6, v = col & 63;
        #pragma unroll
        for (int i = 0; i < 8; ++i) {
          const int r = i + 8*hi;
          PQ[((wv*16 + r)*NB + nb)*PP + v] = f2bf(acc[n][i]);
        }
      }
    } else {                    // gate / res pre-activations
      const int g = wv - 6;
      #pragma unroll
      for (int n = 0; n < 16; ++n) {
        const int col = n*16 + lm;
        #pragma unroll
        for (int i = 0; i < 8; ++i)
          GR[(g*16 + i + 8*hi)*NCOL + col] = f2bf(acc[n][i]);
      }
    }
    __syncthreads();

    // Stage 2a: p_h += q_h @ A_h   (uses q*A2 == (q*A)*A)
    {
      floatx8 sacc[6];
      #pragma unroll
      for (int ii = 0; ii < 6; ++ii) {
        const int tt = wv*6 + ii;
        const int h = tt >> 4, nb = (tt >> 2) & 3, n = tt & 3;
        floatx8 s = vzero;
        #pragma unroll
        for (int kt = 0; kt < 2; ++kt) {
          Frag a, b;
          const unsigned short* qrow = PQ + (((3+h)*16 + lm)*NB + nb)*PP + kt*32 + hi*8;
          a.q[0] = *(const uintx4*)(qrow);
          a.q[1] = *(const uintx4*)(qrow + 16);
          const unsigned short* arow = AS + (h*64 + n*16 + lm)*AP + kt*32 + hi*8;
          b.q[0] = *(const uintx4*)(arow);
          b.q[1] = *(const uintx4*)(arow + 16);
          s = wmma_bf16(a, b, s);
        }
        sacc[ii] = s;
      }
      #pragma unroll
      for (int ii = 0; ii < 6; ++ii) {
        const int tt = wv*6 + ii;
        const int h = tt >> 4, nb = (tt >> 2) & 3, n = tt & 3;
        #pragma unroll
        for (int i = 0; i < 8; ++i) {
          const int m = i + 8*hi;
          const int idx = ((h*16 + m)*NB + nb)*PP + n*16 + lm;
          PQ[idx] = f2bf(bf2f(PQ[idx]) + sacc[ii][i]);
        }
      }
    }
    __syncthreads();

    // Stage 2b: y = sum_h u_h @ A_h ; fused = relu(y*sigmoid(g)+r) -> FS
    {
      #pragma unroll
      for (int ii = 0; ii < 2; ++ii) {
        const int tt = wv*2 + ii;
        const int nb = tt >> 2, n = tt & 3;
        floatx8 y = vzero;
        #pragma unroll
        for (int h = 0; h < 3; ++h) {
          #pragma unroll
          for (int kt = 0; kt < 2; ++kt) {
            Frag a, b;
            const unsigned short* prow = PQ + ((h*16 + lm)*NB + nb)*PP + kt*32 + hi*8;
            a.q[0] = *(const uintx4*)(prow);
            a.q[1] = *(const uintx4*)(prow + 16);
            const unsigned short* arow = AS + (h*64 + n*16 + lm)*AP + kt*32 + hi*8;
            b.q[0] = *(const uintx4*)(arow);
            b.q[1] = *(const uintx4*)(arow + 16);
            y = wmma_bf16(a, b, y);
          }
        }
        #pragma unroll
        for (int i = 0; i < 8; ++i) {
          const int m = i + 8*hi;
          const int w = n*16 + lm;
          const int col = nb*64 + w;
          const float gv = bf2f(GR[m*NCOL + col]);
          const float rv = bf2f(GR[(16 + m)*NCOL + col]);
          const float gate = 1.f / (1.f + __expf(-gv));
          float f = y[i]*gate + rv;
          f = f > 0.f ? f : 0.f;
          FS[(nb*VDIM + w)*FP + (t*16 + m)] = f2bf(f);
        }
      }
    }
    __syncthreads();
  }

  // ---- Stage 3: H = leaky_relu(Wc1 @ fused) -> HS (overlays dead XS) ----
  {
    floatx8 hacc[16];
    #pragma unroll
    for (int n = 0; n < 16; ++n) hacc[n] = vzero;
    #pragma unroll
    for (int kt = 0; kt < 4; ++kt) {
      Frag a;
      const unsigned short* wrow = wc1b + (wv*16 + lm)*CDIM + kt*32 + hi*8;
      a.q[0] = *(const uintx4*)(wrow);
      a.q[1] = *(const uintx4*)(wrow + 16);
      #pragma unroll
      for (int n = 0; n < 16; ++n) {
        Frag b;
        const unsigned short* frow = FS + (n*16 + lm)*FP + kt*32 + hi*8;
        b.q[0] = *(const uintx4*)(frow);
        b.q[1] = *(const uintx4*)(frow + 16);
        hacc[n] = wmma_bf16(a, b, hacc[n]);
      }
    }
    #pragma unroll
    for (int n = 0; n < 16; ++n) {
      const int col = n*16 + lm;
      #pragma unroll
      for (int i = 0; i < 8; ++i) {
        const int o = wv*16 + i + 8*hi;
        float v = hacc[n][i];
        v = v > 0.f ? v : 0.1f*v;
        HS[col*FP + o] = f2bf(v);
      }
    }
  }
  __syncthreads();

  // ---- Stage 4: out = scale * (Wc2 @ H), coalesced f32 stores ----
  {
    floatx8 oacc[16];
    #pragma unroll
    for (int n = 0; n < 16; ++n) oacc[n] = vzero;
    #pragma unroll
    for (int kt = 0; kt < 4; ++kt) {
      Frag a;
      const unsigned short* wrow = wc2b + (wv*16 + lm)*CDIM + kt*32 + hi*8;
      a.q[0] = *(const uintx4*)(wrow);
      a.q[1] = *(const uintx4*)(wrow + 16);
      #pragma unroll
      for (int n = 0; n < 16; ++n) {
        Frag b;
        const unsigned short* hrow = HS + (n*16 + lm)*FP + kt*32 + hi*8;
        b.q[0] = *(const uintx4*)(hrow);
        b.q[1] = *(const uintx4*)(hrow + 16);
        oacc[n] = wmma_bf16(a, b, oacc[n]);
      }
    }
    const float sc = scale[0];
    #pragma unroll
    for (int n = 0; n < 16; ++n) {
      const int col = n*16 + lm;
      const int nb = col >> 6, vv = col & 63;
      #pragma unroll
      for (int i = 0; i < 8; ++i) {
        const int o = wv*16 + i + 8*hi;
        out[((size_t)(b0 + nb)*CDIM + o)*VDIM + vv] = sc * oacc[n][i];
      }
    }
  }
}

extern "C" void kernel_launch(void* const* d_in, const int* in_sizes, int n_in,
                              void* d_out, int out_size, void* d_ws, size_t ws_size,
                              hipStream_t stream) {
  (void)in_sizes; (void)n_in; (void)out_size; (void)ws_size;
  const float* z    = (const float*)d_in[0];
  const float* ve   = (const float*)d_in[1];
  const float* A    = (const float*)d_in[2];
  const float* W1   = (const float*)d_in[3];
  const float* W2   = (const float*)d_in[4];
  const float* Wg   = (const float*)d_in[5];
  const float* Wres = (const float*)d_in[6];
  const float* Wc1  = (const float*)d_in[7];
  const float* Wc2  = (const float*)d_in[8];
  const float* sc   = (const float*)d_in[9];
  unsigned short* ws = (unsigned short*)d_ws;
  float* out = (float*)d_out;

  vc_prep<<<(WS_ELTS + 255)/256, 256, 0, stream>>>(W1, W2, Wg, Wres, Wc1, Wc2, A, ws);

  (void)hipFuncSetAttribute((const void*)vc_main,
                            hipFuncAttributeMaxDynamicSharedMemorySize, SMEM_BYTES);
  vc_main<<<BDIM / NB, 256, SMEM_BYTES, stream>>>(z, ve, ws, sc, out);
}
